// AeloruLayer_31001073942670
// MI455X (gfx1250) — matrix operand order repaired and checked by reading, not compiled, not run on hardware
//
#include <hip/hip_runtime.h>

typedef unsigned short u16;
typedef __attribute__((ext_vector_type(16))) __bf16 v16bf;
typedef __attribute__((ext_vector_type(8)))  float  v8f;
typedef __attribute__((ext_vector_type(4)))  unsigned int u32x4;
typedef __attribute__((ext_vector_type(8)))  int  i32x8;
typedef __attribute__((ext_vector_type(4)))  int  i32x4;

#define D 4096                 // D_IN == D_OUT
#define M_TOTAL 8192           // B*S = 4*2048
#define NELEM 16777216LL       // 4096*4096

union FragBF { uint4 u[2]; v16bf v; };

__device__ __forceinline__ u16 f2bf(float f) {
    unsigned u = __float_as_uint(f);
    unsigned r = (u + 0x7FFFu + ((u >> 16) & 1u)) >> 16;   // RNE
    return (u16)r;
}

// ---------------------------------------------------------------------------
// 1) Per-block partial sums of ||W0||^2 and ||dw||^2 (dw built on the fly)
// ---------------------------------------------------------------------------
__global__ __launch_bounds__(256) void norms_partial(
    const float* __restrict__ W0, const float* __restrict__ lA,
    const float* __restrict__ lB, const float* __restrict__ m,
    const float* __restrict__ F, float* __restrict__ partials)
{
    const int tid = threadIdx.x;
    const long long stride = (long long)gridDim.x * blockDim.x;
    float sW0 = 0.f, sDw = 0.f;
    for (long long idx = (long long)blockIdx.x * blockDim.x + tid; idx < NELEM; idx += stride) {
        const int o = (int)(idx >> 12);
        const int i = (int)(idx & 4095);
        float dot = 0.f;
#pragma unroll
        for (int r = 0; r < 16; ++r)
            dot = fmaf(lB[o * 16 + r], lA[r * D + i], dot);
        const float dw = m[o] * dot / fmaf(10.0f, F[idx], 1.0f);
        const float w0 = W0[idx];
        sW0 = fmaf(w0, w0, sW0);
        sDw = fmaf(dw, dw, sDw);
    }
    __shared__ float red0[256], red1[256];
    red0[tid] = sW0; red1[tid] = sDw;
    __syncthreads();
    for (int s = 128; s > 0; s >>= 1) {
        if (tid < s) { red0[tid] += red0[tid + s]; red1[tid] += red1[tid + s]; }
        __syncthreads();
    }
    if (tid == 0) {
        partials[blockIdx.x * 2 + 0] = red0[0];
        partials[blockIdx.x * 2 + 1] = red1[0];
    }
}

// ---------------------------------------------------------------------------
// 2) Final reduce + energy-budget clamp factor
// ---------------------------------------------------------------------------
__global__ __launch_bounds__(256) void reduce_factor(
    const float* __restrict__ partials, float* __restrict__ factor)
{
    const int tid = threadIdx.x;
    float sW0 = 0.f, sDw = 0.f;
    for (int i = tid; i < 1024; i += 256) {
        sW0 += partials[2 * i + 0];
        sDw += partials[2 * i + 1];
    }
    __shared__ float red0[256], red1[256];
    red0[tid] = sW0; red1[tid] = sDw;
    __syncthreads();
    for (int s = 128; s > 0; s >>= 1) {
        if (tid < s) { red0[tid] += red0[tid + s]; red1[tid] += red1[tid + s]; }
        __syncthreads();
    }
    if (tid == 0) {
        const float w0n = sqrtf(red0[0]);
        const float dwn = sqrtf(red1[0]);
        const float maxA = 0.15f * w0n;
        float f = 1.0f;
        if ((dwn > maxA) && (dwn > 1e-8f)) f = maxA / fmaxf(dwn, 1e-8f);
        *factor = f;
    }
}

// ---------------------------------------------------------------------------
// 3) W_eff = W0 + W_acc + factor * m * (B@A) / (1 + 10*F)  -> bf16
// ---------------------------------------------------------------------------
__global__ __launch_bounds__(256) void build_weff(
    const float* __restrict__ W0, const float* __restrict__ Wacc,
    const float* __restrict__ lA, const float* __restrict__ lB,
    const float* __restrict__ m, const float* __restrict__ F,
    const float* __restrict__ factorp, u16* __restrict__ weff)
{
    const long long id = (long long)blockIdx.x * blockDim.x + threadIdx.x; // 0..2^22-1
    const int o  = (int)(id >> 10);
    const int i0 = (int)(id & 1023) << 2;
    const long long base = (long long)o * D + i0;

    const float fac = *factorp;
    const float4 w0 = *(const float4*)(W0 + base);
    const float4 wa = *(const float4*)(Wacc + base);
    const float4 fm = *(const float4*)(F + base);

    float d0 = 0.f, d1 = 0.f, d2 = 0.f, d3 = 0.f;
#pragma unroll
    for (int r = 0; r < 16; ++r) {
        const float b = lB[o * 16 + r];
        const float4 a = *(const float4*)(lA + r * D + i0);
        d0 = fmaf(b, a.x, d0); d1 = fmaf(b, a.y, d1);
        d2 = fmaf(b, a.z, d2); d3 = fmaf(b, a.w, d3);
    }
    const float s = fac * m[o];
    const float e0 = w0.x + wa.x + s * d0 / fmaf(10.f, fm.x, 1.f);
    const float e1 = w0.y + wa.y + s * d1 / fmaf(10.f, fm.y, 1.f);
    const float e2 = w0.z + wa.z + s * d2 / fmaf(10.f, fm.z, 1.f);
    const float e3 = w0.w + wa.w + s * d3 / fmaf(10.f, fm.w, 1.f);

    uint2 packed;
    packed.x = (unsigned)f2bf(e0) | ((unsigned)f2bf(e1) << 16);
    packed.y = (unsigned)f2bf(e2) | ((unsigned)f2bf(e3) << 16);
    *(uint2*)(weff + base) = packed;
}

// ---------------------------------------------------------------------------
// 4) x fp32 -> bf16
// ---------------------------------------------------------------------------
__global__ __launch_bounds__(256) void cast_x(
    const float* __restrict__ x, u16* __restrict__ xb)
{
    const long long id = (long long)blockIdx.x * blockDim.x + threadIdx.x; // 0..2^23-1
    const float4 v = *(const float4*)(x + id * 4);
    uint2 packed;
    packed.x = (unsigned)f2bf(v.x) | ((unsigned)f2bf(v.y) << 16);
    packed.y = (unsigned)f2bf(v.z) | ((unsigned)f2bf(v.w) << 16);
    *(uint2*)(xb + id * 4) = packed;
}

// ---------------------------------------------------------------------------
// TDM: issue a 2-D tile load (rows x 32 bf16, row stride 4096 elements) from
// global to LDS with hardware padding: 16 DWORDs (64B row) then +4 DWORDs
// (16B) -> 80B LDS pitch (PITCH=40 halves), matching the fragment layout.
// D# packing per CDNA5 ISA sec. 8.3/8.4 (group0/group1; groups 2/3 zero = 2D).
// This toolchain exposes the 6-arg builtin form:
//   (u32x4 g0, i32x8 g1, i32x4 g2, i32x4 g3, i32x8 pad, i32 cpol)
// ---------------------------------------------------------------------------
__device__ __forceinline__ void tdm_load_tile(const u16* g, unsigned ldsOff, int rows) {
    const unsigned long long ga = (unsigned long long)(uintptr_t)g;
    u32x4 g0;
    g0[0] = 1u;                                       // count=1, user mode, no gather
    g0[1] = ldsOff;                                   // lds_addr (bytes)
    g0[2] = (unsigned)(ga & 0xFFFFFFFFu);             // global_addr[31:0]
    g0[3] = (unsigned)((ga >> 32) & 0x01FFFFFFu)      // global_addr[56:32]
          | (2u << 30);                               // type=2 (image)
    i32x8 g1;
    g1[0] = (1 << 16)                                 // data_size = 2 bytes
          | (1 << 20)                                 // pad_enable
          | (3 << 22)                                 // pad_interval: 16 DWORDs
          | (3 << 25);                                // pad_amount: 4 DWORDs
    g1[1] = (int)(4096u << 16);                       // tensor_dim0[15:0]=4096
    g1[2] = (int)((unsigned)rows << 16);              // dim0[31:16]=0 | tensor_dim1[15:0]=rows
    g1[3] = (int)(32u << 16);                         // dim1[31:16]=0 | tile_dim0=32
    g1[4] = rows;                                     // tile_dim1=rows, tile_dim2=0
    g1[5] = 4096;                                     // tensor_dim0_stride[31:0]
    g1[6] = 0;                                        // stride[47:32] | dim1_stride lo
    g1[7] = 0;
    const i32x4 z4 = {0, 0, 0, 0};
    const i32x8 z8 = {0, 0, 0, 0, 0, 0, 0, 0};
    __builtin_amdgcn_tensor_load_to_lds(g0, g1, z4, z4, z8, 0);
}

// ---------------------------------------------------------------------------
// 5) y = x_bf16 @ W_eff_bf16^T + bias   (M=8192, N=4096, K=4096)
//    Block tile 128x256, 8 waves (2x4), wave tile 64x64 (4x4 WMMA tiles),
//    KC=32 double-buffered in LDS, fills done by the Tensor Data Mover.
// ---------------------------------------------------------------------------
#define BM 128
#define BN 256
#define KC 32
#define PITCH 40   // halves per LDS row (80B: conflict-free, 16B aligned)

__global__ __launch_bounds__(256) void gemm_wmma(
    const u16* __restrict__ xb, const u16* __restrict__ wb,
    const float* __restrict__ bias, float* __restrict__ out)
{
    __shared__ u16 As[2][BM * PITCH];   // 2 x 10240 B
    __shared__ u16 Bs[2][BN * PITCH];   // 2 x 20480 B

    const int t    = threadIdx.x;
    const int lane = t & 31;
    const int wv   = t >> 5;
    const int wm   = wv >> 2;       // 0..1  (M direction)
    const int wn   = wv & 3;        // 0..3  (N direction)
    const int mBase = blockIdx.x * BM;
    const int nBase = blockIdx.y * BN;

    const u16* gA = xb + (long long)mBase * D;   // A tile origin (row-major, K contiguous)
    const u16* gB = wb + (long long)nBase * D;   // B tile origin
    const unsigned ldsA0 = (unsigned)(uintptr_t)&As[0][0];
    const unsigned ldsA1 = (unsigned)(uintptr_t)&As[1][0];
    const unsigned ldsB0 = (unsigned)(uintptr_t)&Bs[0][0];
    const unsigned ldsB1 = (unsigned)(uintptr_t)&Bs[1][0];

    v8f acc[4][4];
#pragma unroll
    for (int i = 0; i < 4; ++i)
#pragma unroll
        for (int j = 0; j < 4; ++j)
            acc[i][j] = (v8f){0.f, 0.f, 0.f, 0.f, 0.f, 0.f, 0.f, 0.f};

    // Prologue: wave 0 issues the TDM fills for k-chunk 0 into buffer 0.
    if (wv == 0) {
        tdm_load_tile(gA, ldsA0, BM);
        tdm_load_tile(gB, ldsB0, BN);
    }

    const int half   = lane >> 4;       // 0 / 1
    const int laneMN = lane & 15;

    const int NK = D / KC;              // 128
    for (int kc = 0; kc < NK; ++kc) {
        const int buf = kc & 1;
        if (wv == 0) __builtin_amdgcn_s_wait_tensorcnt(0);  // chunk kc landed in LDS
        __syncthreads();

        if ((wv == 0) && (kc + 1 < NK)) {   // TDM prefetch of chunk kc+1 -> other buffer
            const long long ko = (long long)(kc + 1) * KC;
            tdm_load_tile(gA + ko, buf ? ldsA0 : ldsA1, BM);
            tdm_load_tile(gB + ko, buf ? ldsB0 : ldsB1, BN);
        }

        // A fragments: lane<16 -> row M=lane, K {0..7,16..23}; lane>=16 -> K {8..15,24..31}
        FragBF a[4];
#pragma unroll
        for (int ti = 0; ti < 4; ++ti) {
            const int row = wm * 64 + ti * 16 + laneMN;
            const u16* p = &As[buf][row * PITCH + half * 8];
            a[ti].u[0] = *(const uint4*)p;
            a[ti].u[1] = *(const uint4*)(p + 16);
        }
        // B fragments: lane<16 -> col N=lane, K {0..7,16..23}; lane>=16 -> K {8..15,24..31}
        FragBF b[4];
#pragma unroll
        for (int tj = 0; tj < 4; ++tj) {
            const int col = wn * 64 + tj * 16 + laneMN;
            const u16* p = &Bs[buf][col * PITCH + half * 8];
            b[tj].u[0] = *(const uint4*)p;
            b[tj].u[1] = *(const uint4*)(p + 16);
        }

#pragma unroll
        for (int ti = 0; ti < 4; ++ti)
#pragma unroll
            for (int tj = 0; tj < 4; ++tj)
                acc[ti][tj] = __builtin_amdgcn_wmma_f32_16x16x32_bf16(
                    false, a[ti].v, false, b[tj].v,
                    (short)0, acc[ti][tj], false, false);
    }

    // Epilogue: C/D layout -> lanes 0-15: M=v, N=lane; lanes 16-31: M=v+8, N=lane-16
    const int rOff = half * 8;
#pragma unroll
    for (int ti = 0; ti < 4; ++ti) {
#pragma unroll
        for (int tj = 0; tj < 4; ++tj) {
            const int n  = nBase + wn * 64 + tj * 16 + laneMN;
            const float bv = bias[n];
#pragma unroll
            for (int v = 0; v < 8; ++v) {
                const int mg = mBase + wm * 64 + ti * 16 + rOff + v;
                out[(long long)mg * D + n] = acc[ti][tj][v] + bv;
            }
        }
    }
}

// ---------------------------------------------------------------------------
extern "C" void kernel_launch(void* const* d_in, const int* in_sizes, int n_in,
                              void* d_out, int out_size, void* d_ws, size_t ws_size,
                              hipStream_t stream) {
    const float* x    = (const float*)d_in[0];
    const float* W0   = (const float*)d_in[1];
    const float* Wacc = (const float*)d_in[2];
    const float* bias = (const float*)d_in[3];
    const float* lA   = (const float*)d_in[4];
    const float* lB   = (const float*)d_in[5];
    const float* m    = (const float*)d_in[6];
    const float* F    = (const float*)d_in[7];
    float* out = (float*)d_out;

    // Workspace layout (bytes):
    //   [0, 32MB)      W_eff bf16  (4096*4096*2)
    //   [32MB, 96MB)   x bf16      (8192*4096*2)
    //   [96MB, +8KB)   per-block norm partials (1024*2 floats)
    //   then           clamp factor (1 float)
    u16*   weff     = (u16*)d_ws;
    u16*   xb       = (u16*)((char*)d_ws + (size_t)33554432);
    float* partials = (float*)((char*)d_ws + (size_t)100663296);
    float* factorp  = partials + 2048;

    norms_partial<<<1024, 256, 0, stream>>>(W0, lA, lB, m, F, partials);
    reduce_factor<<<1, 256, 0, stream>>>(partials, factorp);
    build_weff<<<16384, 256, 0, stream>>>(W0, Wacc, lA, lB, m, F, factorp, weff);
    cast_x<<<32768, 256, 0, stream>>>(x, xb);

    dim3 grid(M_TOTAL / BM, D / BN);   // 64 x 16
    gemm_wmma<<<grid, 256, 0, stream>>>(xb, weff, bias, out);
}
